// MultiHeadedAttention_35167192220400
// MI455X (gfx1250) — compile-verified, matrix operand order checked by reading, not run on hardware
//
#include <hip/hip_runtime.h>

typedef __attribute__((ext_vector_type(16))) _Float16 v16h;
typedef __attribute__((ext_vector_type(8)))  _Float16 v8h;
typedef __attribute__((ext_vector_type(8)))  float    v8f;
typedef __attribute__((ext_vector_type(4)))  float    v4f;

constexpr int Bb  = 2;
constexpr int T   = 2048;
constexpr int D   = 1024;
constexpr int H   = 16;
constexpr int DK  = 64;
constexpr int M   = Bb * T;      // 4096 rows
constexpr int WIN = 64;
constexpr int MAXD = 128;
constexpr float NEG = -1.0e9f;

constexpr int KSTEP = 64;        // K per LDS stage (2 WMMA K-chunks)
constexpr int LDSK  = 72;        // padded LDS row stride (f16), 16B multiple

// ---- TDM availability -------------------------------------------------------
#if defined(__has_builtin)
#if __has_builtin(__builtin_amdgcn_tensor_load_to_lds) && \
    __has_builtin(__builtin_amdgcn_s_wait_tensorcnt)
#define USE_TDM 1
#endif
#endif
#ifndef USE_TDM
#define USE_TDM 0
#endif

// ---- fragment helpers -------------------------------------------------------
// A-frag (16x32, M=lane%16) and B-frag (32x16, N=lane%16) share the per-lane
// pattern: 8 contiguous K f16 at k0=(lane/16)*8 and 8 more at k0+16.
__device__ __forceinline__ v16h load_frag(const _Float16* rowp, int k0) {
  v8h lo = *(const v8h*)(rowp + k0);
  v8h hi = *(const v8h*)(rowp + k0 + 16);
  v16h r;
#pragma unroll
  for (int i = 0; i < 8; ++i) { r[i] = lo[i]; r[i + 8] = hi[i]; }
  return r;
}

__device__ __forceinline__ v8f wmma_f16(v16h a, v16h b, v8f c) {
  return __builtin_amdgcn_wmma_f32_16x16x32_f16(false, a, false, b, (short)0, c,
                                                false, false);
}

// ---- conversion passes ------------------------------------------------------
__global__ void __launch_bounds__(256)
cvt_kernel(const float* __restrict__ in, _Float16* __restrict__ out, int n8) {
  int i = blockIdx.x * 256 + threadIdx.x;
  if (i >= n8) return;
  const v4f* p = (const v4f*)in + (size_t)i * 2;
  v4f a = p[0], b = p[1];
  v8h o;
  o[0] = (_Float16)a.x; o[1] = (_Float16)a.y; o[2] = (_Float16)a.z; o[3] = (_Float16)a.w;
  o[4] = (_Float16)b.x; o[5] = (_Float16)b.y; o[6] = (_Float16)b.z; o[7] = (_Float16)b.w;
  *((v8h*)out + i) = o;
}

// Wt[n][k] = (f16) W[k][n]  (W is only 4 MB: lives in L2, strided reads cheap)
__global__ void __launch_bounds__(256)
cvtT_kernel(const float* __restrict__ in, _Float16* __restrict__ out) {
  int idx = blockIdx.x * 256 + threadIdx.x;     // over D*D
  int nr = idx >> 10, k = idx & (D - 1);
  out[idx] = (_Float16)in[(size_t)k * D + nr];
}

// ---- Tensor Data Mover tile load -------------------------------------------
#if USE_TDM
typedef __attribute__((ext_vector_type(4))) unsigned int u32x4;
typedef __attribute__((ext_vector_type(4))) int i32x4;
typedef __attribute__((ext_vector_type(8))) int i32x8;

// 2-D f16 tile: KSTEP contiguous elems per row, `tile_rows` rows, source row
// stride D elems.  LDS pad: after every 32 dwords (=64 f16 row) insert 4 dwords
// (=8 f16) -> LDS row stride 72 f16 = LDSK.
__device__ __forceinline__ void tdm_load_tile(unsigned lds_addr,
                                              const _Float16* gptr,
                                              int tensor_rows, int tile_rows) {
  unsigned long long ga = (unsigned long long)(size_t)gptr;
  u32x4 g0 = {0u, 0u, 0u, 0u};
  g0[0] = 1u;                                              // count=1 user desc
  g0[1] = lds_addr;                                        // LDS byte address
  g0[2] = (unsigned)ga;                                    // global addr lo
  g0[3] = (unsigned)((ga >> 32) & 0x01FFFFFFull) | (2u << 30);  // hi + type=2
  i32x8 g1 = {0, 0, 0, 0, 0, 0, 0, 0};
  g1[0] = (int)((1u << 16)        // data_size = 2 bytes
              | (1u << 20)        // pad_enable
              | (4u << 22)        // pad_interval = 32 dwords
              | (3u << 25));      // pad_amount  = 4 dwords
  g1[1] = (int)(((unsigned)D & 0xFFFFu) << 16);            // tensor_dim0 lo16
  g1[2] = (int)(((unsigned)D >> 16) |
                (((unsigned)tensor_rows & 0xFFFFu) << 16)); // dim0 hi / dim1 lo
  g1[3] = (int)(((unsigned)tensor_rows >> 16) |
                ((unsigned)KSTEP << 16));                   // dim1 hi / tile_dim0
  g1[4] = (int)((unsigned)tile_rows & 0xFFFFu);             // tile_dim1, dim2=0
  g1[5] = (int)D;                                           // dim0_stride lo32
  i32x4 z4 = {0, 0, 0, 0};
#if __clang_major__ >= 23
  i32x8 z8 = {0, 0, 0, 0, 0, 0, 0, 0};
  __builtin_amdgcn_tensor_load_to_lds(g0, g1, z4, z4, z8, 0);
#else
  __builtin_amdgcn_tensor_load_to_lds(g0, g1, z4, z4, 0);
#endif
}
#endif

// one K-stage of the wave's 32x64 tile: 16 WMMA from given LDS buffers
__device__ __forceinline__ void mm_stage(const _Float16 (*sAb)[LDSK],
                                         const _Float16 (*sBb)[LDSK],
                                         int wv, int r, int k0h,
                                         v8f (&acc)[2][4]) {
#pragma unroll
  for (int kc = 0; kc < KSTEP; kc += 32) {
    v16h a0 = load_frag(&sAb[wv * 32 + r][kc], k0h);
    v16h a1 = load_frag(&sAb[wv * 32 + 16 + r][kc], k0h);
#pragma unroll
    for (int nt = 0; nt < 4; ++nt) {
      v16h bf = load_frag(&sBb[nt * 16 + r][kc], k0h);
      acc[0][nt] = wmma_f16(a0, bf, acc[0][nt]);
      acc[1][nt] = wmma_f16(a1, bf, acc[1][nt]);
    }
  }
}

// ---- f16 GEMM ---------------------------------------------------------------
// C[4096x1024] = A[4096xK,f16] * Wt^T (Wt is [n][k], f16) + bias.
// 128 threads = 4 waves, block tile 128x64, wave tile 32x64.  TDM double-
// buffered pipeline: wave 0 issues next tile's 2 DMA ops, waits tensorcnt<=2
// (in-order => current tile ready), barrier, all compute.
// MODE 0: f16 out[(b*H+h)*T + t][dk]   (Q/K layout)
// MODE 1: f16 out[(b*H+h)*DK + dk][t]  (V transposed)
// MODE 2: f32 out[m][n]                (final output)
template <int MODE>
__global__ void __launch_bounds__(128)
gemm16(const _Float16* __restrict__ A, const _Float16* __restrict__ Wt,
       const float* __restrict__ bias, _Float16* __restrict__ outH,
       float* __restrict__ outF) {
  const int tid  = threadIdx.x;
  const int lane = tid & 31;
  const int wv   = tid >> 5;
  const int m0   = blockIdx.x * 128;
  const int n0   = blockIdx.y * 64;
  const int r    = lane & 15;
  const int k0h  = (lane >> 4) * 8;

  v8f acc[2][4];
#pragma unroll
  for (int mt = 0; mt < 2; ++mt)
#pragma unroll
    for (int nt = 0; nt < 4; ++nt)
#pragma unroll
      for (int j = 0; j < 8; ++j) acc[mt][nt][j] = 0.f;

  const _Float16* Abase = A + (size_t)m0 * D;
  const _Float16* Bbase = Wt + (size_t)n0 * D;

#if USE_TDM
  __shared__ alignas(16) _Float16 sA0[128][LDSK], sA1[128][LDSK];
  __shared__ alignas(16) _Float16 sB0[64][LDSK], sB1[64][LDSK];
  const unsigned lA0 = (unsigned)(size_t)(void*)&sA0[0][0];
  const unsigned lA1 = (unsigned)(size_t)(void*)&sA1[0][0];
  const unsigned lB0 = (unsigned)(size_t)(void*)&sB0[0][0];
  const unsigned lB1 = (unsigned)(size_t)(void*)&sB1[0][0];

  if (wv == 0) {                       // prologue: fill buffer 0
    tdm_load_tile(lA0, Abase, M, 128);
    tdm_load_tile(lB0, Bbase, D, 64);
  }
  for (int kb = 0; kb < D; kb += 2 * KSTEP) {
    // ---- stage A: compute buf0, prefetch buf1 (kb+KSTEP < D always) --------
    __syncthreads();                   // everyone done reading buf1 (prev iter)
    if (wv == 0) {
      tdm_load_tile(lA1, Abase + kb + KSTEP, M, 128);
      tdm_load_tile(lB1, Bbase + kb + KSTEP, D, 64);
      __builtin_amdgcn_s_wait_tensorcnt(2);   // buf0 ready
    }
    __syncthreads();
    mm_stage(sA0, sB0, wv, r, k0h, acc);
    // ---- stage B: compute buf1, prefetch buf0 ------------------------------
    __syncthreads();
    if (wv == 0) {
      if (kb + 2 * KSTEP < D) {
        tdm_load_tile(lA0, Abase + kb + 2 * KSTEP, M, 128);
        tdm_load_tile(lB0, Bbase + kb + 2 * KSTEP, D, 64);
        __builtin_amdgcn_s_wait_tensorcnt(2); // buf1 ready
      } else {
        __builtin_amdgcn_s_wait_tensorcnt(0);
      }
    }
    __syncthreads();
    mm_stage(sA1, sB1, wv, r, k0h, acc);
  }
#else
  __shared__ alignas(16) _Float16 sA0[128][LDSK];
  __shared__ alignas(16) _Float16 sB0[64][LDSK];
  for (int kb = 0; kb < D; kb += KSTEP) {
    __syncthreads();
#pragma unroll
    for (int i = 0; i < 8; ++i) {                       // A tile 128x64 f16
      int e = i * 1024 + tid * 8;
      int row = e >> 6, col = e & 63;
      *(v8h*)&sA0[row][col] = *(const v8h*)(Abase + (size_t)row * D + kb + col);
    }
#pragma unroll
    for (int i = 0; i < 4; ++i) {                       // B tile 64x64 f16
      int e = i * 1024 + tid * 8;
      int row = e >> 6, col = e & 63;
      *(v8h*)&sB0[row][col] = *(const v8h*)(Bbase + (size_t)row * D + kb + col);
    }
    __syncthreads();
    mm_stage(sA0, sB0, wv, r, k0h, acc);
  }
#endif

  // epilogue: C layout element (m = rbase+j, n = lane%16)
  const int rbase = (lane >> 4) * 8;
  const int c     = lane & 15;
#pragma unroll
  for (int mt = 0; mt < 2; ++mt)
#pragma unroll
    for (int nt = 0; nt < 4; ++nt) {
      const int col = n0 + nt * 16 + c;
      const float bcol = bias[col];
      if (MODE == 1) {
        // V^T layout: consecutive j -> consecutive t: pack one b128 store
        v8h pk;
#pragma unroll
        for (int j = 0; j < 8; ++j) pk[j] = (_Float16)(acc[mt][nt][j] + bcol);
        const int mb = m0 + wv * 32 + mt * 16 + rbase;  // 8-aligned, same b
        const int b  = mb >> 11, t = mb & (T - 1);
        const int bh = b * H + (col >> 6);
        *(v8h*)(outH + ((size_t)bh * DK + (col & 63)) * T + t) = pk;
      } else {
#pragma unroll
        for (int j = 0; j < 8; ++j) {
          const int m = m0 + wv * 32 + mt * 16 + rbase + j;
          const float val = acc[mt][nt][j] + bcol;
          if (MODE == 2) {
            outF[(size_t)m * D + col] = val;
          } else {
            const int b  = m >> 11;
            const int t  = m & (T - 1);
            const int bh = b * H + (col >> 6);
            outH[((size_t)bh * T + t) * DK + (col & 63)] = (_Float16)val;
          }
        }
      }
    }
}

// ---- banded attention -------------------------------------------------------
// One wave per (b,h, 16-query tile); 9 key tiles of 16 cover |k-q|<=64 exactly.
__global__ void __launch_bounds__(32)
attn_kernel(const _Float16* __restrict__ qw, const _Float16* __restrict__ kw,
            const _Float16* __restrict__ vt, const float* __restrict__ rel,
            _Float16* __restrict__ outA) {
  __shared__ alignas(16) _Float16 sP[16][160];
  __shared__ float sRel[257];

  const int lane  = threadIdx.x;
  const int nq    = T / 16;
  const int qt    = blockIdx.x % nq;
  const int bh    = blockIdx.x / nq;
  const int hh    = bh % H;
  const int bb    = bh / H;
  const int q0    = qt * 16;
  const int r     = lane & 15;
  const int k0h   = (lane >> 4) * 8;
  const int rbase = (lane >> 4) * 8;
  const int c     = lane & 15;

  // stage this head's bias column (257 floats) into LDS
  for (int i = lane; i < 257; i += 32) sRel[i] = rel[(size_t)i * H + hh];

  const _Float16* qrow = qw + ((size_t)bh * T + q0 + r) * DK;
  v16h aq0 = load_frag(qrow, k0h);
  v16h aq1 = load_frag(qrow + 32, k0h);

  v8f sc[9];
#pragma unroll
  for (int t = 0; t < 9; ++t) {
#pragma unroll
    for (int j = 0; j < 8; ++j) sc[t][j] = 0.f;
    int key = q0 - 64 + t * 16 + r;                       // this lane's column
    int kc  = key < 0 ? 0 : (key > T - 1 ? T - 1 : key);  // safe addr
    const _Float16* krow = kw + ((size_t)bh * T + kc) * DK;
    v16h b0 = load_frag(krow, k0h);
    v16h b1 = load_frag(krow + 32, k0h);
    sc[t] = wmma_f16(aq0, b0, sc[t]);
    sc[t] = wmma_f16(aq1, b1, sc[t]);
  }
  __syncthreads();                                        // sRel visible

  constexpr float scale = 0.125f;  // 1/sqrt(64)
#pragma unroll
  for (int t = 0; t < 9; ++t) {
    const int key = q0 - 64 + t * 16 + c;
    const bool keyok = (unsigned)key < (unsigned)T;
#pragma unroll
    for (int j = 0; j < 8; ++j) {
      const int d = key - (q0 + rbase + j);     // |d| <= 79 < MAXD: no clip
      const bool dok = (unsigned)(d + WIN) <= 2u * WIN;
      const float bias = sRel[d + MAXD];
      sc[t][j] = (keyok && dok) ? (sc[t][j] * scale + bias) : NEG;
    }
  }

  // softmax per row (row j lives across lanes 0..15 / 16..31)
#pragma unroll
  for (int j = 0; j < 8; ++j) {
    float mx = NEG;
#pragma unroll
    for (int t = 0; t < 9; ++t) mx = fmaxf(mx, sc[t][j]);
#pragma unroll
    for (int off = 1; off < 16; off <<= 1) mx = fmaxf(mx, __shfl_xor(mx, off, 16));
    float s = 0.f;
#pragma unroll
    for (int t = 0; t < 9; ++t) {
      float p = __expf(sc[t][j] - mx);
      sc[t][j] = p;
      s += p;
    }
#pragma unroll
    for (int off = 1; off < 16; off <<= 1) s += __shfl_xor(s, off, 16);
    const float inv = 1.f / s;
#pragma unroll
    for (int t = 0; t < 9; ++t) sc[t][j] *= inv;
  }

  // C-layout -> A-layout via LDS (zero-pad cols 144..159)
#pragma unroll
  for (int t = 0; t < 9; ++t)
#pragma unroll
    for (int j = 0; j < 8; ++j) sP[rbase + j][t * 16 + c] = (_Float16)sc[t][j];
#pragma unroll
  for (int j = 0; j < 8; ++j) sP[rbase + j][144 + c] = (_Float16)0.f;
  __syncthreads();

  v8f o[4];
#pragma unroll
  for (int nt = 0; nt < 4; ++nt)
#pragma unroll
    for (int j = 0; j < 8; ++j) o[nt][j] = 0.f;

  const int s0 = q0 - 64;
#pragma unroll
  for (int kk = 0; kk < 5; ++kk) {
    v16h ap = load_frag(&sP[r][kk * 32], k0h);
    int g1 = s0 + kk * 32 + k0h;      // 8-aligned key groups: clamp whole group
    int g2 = g1 + 16;                 // (P=0 wherever clamped)
    g1 = g1 < 0 ? 0 : (g1 > T - 8 ? T - 8 : g1);
    g2 = g2 < 0 ? 0 : (g2 > T - 8 ? T - 8 : g2);
#pragma unroll
    for (int nt = 0; nt < 4; ++nt) {
      const _Float16* vrow = vt + ((size_t)bh * DK + nt * 16 + r) * T;
      v8h lo = *(const v8h*)(vrow + g1);
      v8h hi = *(const v8h*)(vrow + g2);
      v16h bf;
#pragma unroll
      for (int i = 0; i < 8; ++i) { bf[i] = lo[i]; bf[i + 8] = hi[i]; }
      o[nt] = wmma_f16(ap, bf, o[nt]);
    }
  }

  // transpose result through sP, then 4 coalesced b128 stores per lane
  __syncthreads();
#pragma unroll
  for (int nt = 0; nt < 4; ++nt)
#pragma unroll
    for (int j = 0; j < 8; ++j)
      sP[rbase + j][nt * 16 + c] = (_Float16)o[nt][j];
  __syncthreads();
  {
    const int row = lane & 15, half = lane >> 4;
    _Float16* orow =
        outA + ((size_t)(bb * T + q0 + row)) * D + hh * DK + half * 32;
#pragma unroll
    for (int i = 0; i < 4; ++i)
      *(v8h*)(orow + i * 8) = *(const v8h*)&sP[row][half * 32 + i * 8];
  }
}

// ---- launch -----------------------------------------------------------------
extern "C" void kernel_launch(void* const* d_in, const int* in_sizes, int n_in,
                              void* d_out, int out_size, void* d_ws,
                              size_t ws_size, hipStream_t stream) {
  (void)in_sizes; (void)n_in; (void)out_size; (void)ws_size;
  const float* q_in = (const float*)d_in[0];
  const float* k_in = (const float*)d_in[1];
  const float* v_in = (const float*)d_in[2];
  const float* Wq = (const float*)d_in[3];
  const float* bq = (const float*)d_in[4];
  const float* Wk = (const float*)d_in[5];
  const float* bk = (const float*)d_in[6];
  const float* Wv = (const float*)d_in[7];
  const float* bv = (const float*)d_in[8];
  const float* Wo = (const float*)d_in[9];
  const float* bo = (const float*)d_in[10];
  const float* rel = (const float*)d_in[11];

  char* ws = (char*)d_ws;
  const size_t MB = (size_t)1 << 20;
  _Float16* xq  = (_Float16*)(ws + 0 * MB);    // f16 inputs (8 MB each)
  _Float16* xk  = (_Float16*)(ws + 8 * MB);
  _Float16* xv  = (_Float16*)(ws + 16 * MB);
  _Float16* wqt = (_Float16*)(ws + 24 * MB);   // f16 transposed weights (2 MB)
  _Float16* wkt = (_Float16*)(ws + 26 * MB);
  _Float16* wvt = (_Float16*)(ws + 28 * MB);
  _Float16* wot = (_Float16*)(ws + 30 * MB);
  _Float16* q16 = (_Float16*)(ws + 32 * MB);   // projected Q/K/V (8 MB each)
  _Float16* k16 = (_Float16*)(ws + 40 * MB);
  _Float16* v16 = (_Float16*)(ws + 48 * MB);
  _Float16* at16 = (_Float16*)(ws + 56 * MB);  // attention out (8 MB)

  const int nBT = M * D;                       // 4M elements
  cvt_kernel<<<dim3(nBT / 8 / 256), dim3(256), 0, stream>>>(q_in, xq, nBT / 8);
  cvt_kernel<<<dim3(nBT / 8 / 256), dim3(256), 0, stream>>>(k_in, xk, nBT / 8);
  cvt_kernel<<<dim3(nBT / 8 / 256), dim3(256), 0, stream>>>(v_in, xv, nBT / 8);
  cvtT_kernel<<<dim3(D * D / 256), dim3(256), 0, stream>>>(Wq, wqt);
  cvtT_kernel<<<dim3(D * D / 256), dim3(256), 0, stream>>>(Wk, wkt);
  cvtT_kernel<<<dim3(D * D / 256), dim3(256), 0, stream>>>(Wv, wvt);
  cvtT_kernel<<<dim3(D * D / 256), dim3(256), 0, stream>>>(Wo, wot);

  dim3 grid(M / 128, D / 64), blk(128);
  gemm16<0><<<grid, blk, 0, stream>>>(xq, wqt, bq, q16, nullptr);
  gemm16<0><<<grid, blk, 0, stream>>>(xk, wkt, bk, k16, nullptr);
  gemm16<1><<<grid, blk, 0, stream>>>(xv, wvt, bv, v16, nullptr);
  attn_kernel<<<dim3(Bb * H * (T / 16)), dim3(32), 0, stream>>>(q16, k16, v16,
                                                                rel, at16);
  gemm16<2><<<grid, blk, 0, stream>>>(at16, wot, bo, nullptr, (float*)d_out);
}
